// SetAbstraction_58110907514886
// MI455X (gfx1250) — matrix operand order repaired and checked by reading, not compiled, not run on hardware
//
#include <hip/hip_runtime.h>

typedef _Float16 f16;
typedef __attribute__((ext_vector_type(16))) _Float16 v16h;
typedef __attribute__((ext_vector_type(8)))  _Float16 v8h;
typedef __attribute__((ext_vector_type(8)))  float    v8f;

#define B_    8
#define N_    8192
#define S_    2048       // NPOINT
#define K_    32         // NSAMPLE
#define RAD2  0.04f      // RADIUS^2
#define M_TOT (B_*S_*K_) // 524288 rows of the GEMMs
#define EPSV  1e-5f

// ---------------------------------------------------------------------------
// 1. Farthest point sampling: one workgroup per batch, points in registers,
//    shfl + LDS argmax reduction; 2048 sequential iterations.
// ---------------------------------------------------------------------------
__global__ __launch_bounds__(1024) void fps_kernel(const float* __restrict__ xyz,
                                                   int* __restrict__ fidx)
{
    int b = blockIdx.x;
    const float* X = xyz + (size_t)b * 3 * N_;
    const float* Y = X + N_;
    const float* Z = X + 2 * N_;
    int t = threadIdx.x;
    int lane = t & 31, wv = t >> 5;
    const int PPT = N_ / 1024; // 8 points per thread

    float px[PPT], py[PPT], pz[PPT], pd[PPT];
    for (int i = 0; i < PPT; ++i) {
        int p = t + i * 1024;
        px[i] = X[p]; py[i] = Y[p]; pz[i] = Z[p];
        pd[i] = 1e10f;
    }

    __shared__ float wbest[32];
    __shared__ int   widx[32];
    __shared__ int   sfar;

    int far = 0;
    for (int it = 0; it < S_; ++it) {
        if (t == 0) fidx[b * S_ + it] = far;
        float cx = X[far], cy = Y[far], cz = Z[far];

        float bd = -1.0f; int bi = 0;
        for (int i = 0; i < PPT; ++i) {
            float dx = px[i] - cx, dy = py[i] - cy, dz = pz[i] - cz;
            float d = dx * dx + dy * dy + dz * dz;
            pd[i] = fminf(pd[i], d);
            if (pd[i] > bd) { bd = pd[i]; bi = t + i * 1024; }
        }
        // wave32 butterfly argmax (ties -> smaller index, matching jnp.argmax)
        for (int off = 16; off > 0; off >>= 1) {
            float od = __shfl_xor(bd, off, 32);
            int   oi = __shfl_xor(bi, off, 32);
            if (od > bd || (od == bd && oi < bi)) { bd = od; bi = oi; }
        }
        if (lane == 0) { wbest[wv] = bd; widx[wv] = bi; }
        __syncthreads();
        if (wv == 0) {
            bd = wbest[lane]; bi = widx[lane];
            for (int off = 16; off > 0; off >>= 1) {
                float od = __shfl_xor(bd, off, 32);
                int   oi = __shfl_xor(bi, off, 32);
                if (od > bd || (od == bd && oi < bi)) { bd = od; bi = oi; }
            }
            if (lane == 0) sfar = bi;
        }
        __syncthreads();
        far = sfar;
    }
}

// ---------------------------------------------------------------------------
// 2. Gather sampled centers; also writes the transposed new_xyz output.
// ---------------------------------------------------------------------------
__global__ void gather_newxyz_kernel(const float* __restrict__ xyz,
                                     const int* __restrict__ fidx,
                                     float* __restrict__ newxyz,
                                     float* __restrict__ out)
{
    int q = blockIdx.x * blockDim.x + threadIdx.x;
    if (q >= B_ * S_) return;
    int b = q / S_, s = q % S_;
    int j = fidx[q];
    const float* X = xyz + (size_t)b * 3 * N_;
    for (int c = 0; c < 3; ++c) {
        float v = X[c * N_ + j];
        newxyz[q * 3 + c] = v;
        out[(size_t)b * 3 * S_ + c * S_ + s] = v;
    }
}

// ---------------------------------------------------------------------------
// 3. Ball query: one wave per query, ballot-based in-order compaction of the
//    first 32 in-radius indices; early exit when full.
// ---------------------------------------------------------------------------
__global__ __launch_bounds__(256) void ballquery_kernel(const float* __restrict__ xyz,
                                                        const float* __restrict__ newxyz,
                                                        int* __restrict__ nidx)
{
    int wv = threadIdx.x >> 5, lane = threadIdx.x & 31;
    int q = blockIdx.x * 8 + wv;            // < B_*S_
    int b = q / S_;
    const float* X = xyz + (size_t)b * 3 * N_;
    const float* Y = X + N_;
    const float* Z = X + 2 * N_;
    float qx = newxyz[q * 3 + 0], qy = newxyz[q * 3 + 1], qz = newxyz[q * 3 + 2];

    int cnt = 0, first = -1;
    int* out = nidx + (size_t)q * K_;
    for (int base = 0; base < N_ && cnt < K_; base += 32) {
        int n = base + lane;
        float dx = X[n] - qx, dy = Y[n] - qy, dz = Z[n] - qz;
        bool ok = (dx * dx + dy * dy + dz * dz) <= RAD2;
        unsigned m = (unsigned)__ballot(ok);
        if (first < 0 && m) first = base + (__ffs(m) - 1);
        int pos = cnt + __popc(m & ((1u << lane) - 1u));
        if (ok && pos < K_) out[pos] = n;
        cnt += __popc(m);
        if (cnt > K_) cnt = K_;
    }
    if (first < 0) first = 0;
    if (lane >= cnt) out[lane] = first;   // pad with first neighbor
}

// ---------------------------------------------------------------------------
// 4. Build grouped features (B,S,K,9) as f16, channel-padded to 16.
// ---------------------------------------------------------------------------
__global__ void build_feat_kernel(const float* __restrict__ xyz,
                                  const float* __restrict__ points,
                                  const float* __restrict__ newxyz,
                                  const int* __restrict__ nidx,
                                  f16* __restrict__ feat)
{
    int m = blockIdx.x * blockDim.x + threadIdx.x;   // < M_TOT
    int q = m >> 5;                                   // b*S+s
    int b = q / S_;
    int j = nidx[m];
    const float* X = xyz + (size_t)b * 3 * N_;
    const float* P = points + (size_t)b * 6 * N_;
    f16* f = feat + (size_t)m * 16;
    f[0] = (f16)(X[j]          - newxyz[q * 3 + 0]);
    f[1] = (f16)(X[N_ + j]     - newxyz[q * 3 + 1]);
    f[2] = (f16)(X[2 * N_ + j] - newxyz[q * 3 + 2]);
    for (int c = 0; c < 6; ++c) f[3 + c] = (f16)P[c * N_ + j];
    for (int c = 9; c < 16; ++c) f[c] = (f16)0.f;
}

// ---------------------------------------------------------------------------
// 5. Transpose + pad weights to f16 wT[c][o] (B-matrix friendly layout).
// ---------------------------------------------------------------------------
__global__ void prep_wt_kernel(const float* __restrict__ w, f16* __restrict__ wT,
                               int O, int CIN, int CPAD)
{
    int i = blockIdx.x * blockDim.x + threadIdx.x;
    if (i >= CPAD * O) return;
    int c = i / O, o = i - c * O;
    wT[i] = (c < CIN) ? (f16)w[o * CIN + c] : (f16)0.f;
}

__global__ void zero_kernel(float* __restrict__ p, int n)
{
    int i = blockIdx.x * blockDim.x + threadIdx.x;
    if (i < n) p[i] = 0.f;
}

// ---------------------------------------------------------------------------
// 6. WMMA GEMM: h[m,o] = relu(bn(A))[m,c] * wT[c,o], f32 accumulate,
//    f16 output + per-channel sum/sumsq for the next BN.
//    CIN: stored A channels (= A stride). O: outputs. BN: apply scale/shift+relu
//    to A on load (BN of the *previous* layer, fused).
//    Block = 256 threads = 8 waves; wave owns one 16-row M tile.
// ---------------------------------------------------------------------------
template <int CIN, int O, bool BN>
__global__ __launch_bounds__(256) void gemm_wmma_kernel(const f16* __restrict__ A,
                                                        const f16* __restrict__ wT,
                                                        const float* __restrict__ scale,
                                                        const float* __restrict__ shift,
                                                        f16* __restrict__ Hout,
                                                        float* __restrict__ gsum,
                                                        float* __restrict__ gsq)
{
    constexpr int KBLK = (CIN + 31) / 32;   // k-steps of 32
    constexpr int KTOT = KBLK * 32;

    __shared__ f16   swT[KTOT * O];         // zero-padded weights, lane=K layout
    __shared__ float ssum[O], ssq[O];
    __shared__ float ssc[CIN], ssh[CIN];

    int t = threadIdx.x;
    for (int i = t; i < KTOT * O; i += 256) {
        int c = i / O, o = i - c * O;
        swT[i] = (c < CIN) ? wT[c * O + o] : (f16)0.f;
    }
    if (t < O) { ssum[t] = 0.f; ssq[t] = 0.f; }
    if (BN && t < CIN) { ssc[t] = scale[t]; ssh[t] = shift[t]; }
    __syncthreads();

    int lane = t & 31, wv = t >> 5;
    int m0 = (blockIdx.x * 8 + wv) * 16;
    const f16* Arow = A + (size_t)(m0 + (lane & 15)) * CIN;
    int hsel = (lane >> 4) << 3;            // 0 for lanes 0-15, 8 for 16-31
    __builtin_prefetch(Arow + 128 * CIN, 0, 1);

    // A fragments: ISA 16-bit A 16x32 layout.
    // halfs 0..7  : K = kk + hsel + j ;  halfs 8..15 : K = kk + 16 + hsel + j
    v16h afrag[KBLK];
    for (int kb = 0; kb < KBLK; ++kb) {
        int kk = kb * 32;
        v16h a;
#pragma unroll
        for (int j = 0; j < 8; ++j) {
            int c0 = kk + hsel + j;
            int c1 = kk + 16 + hsel + j;
            float x0 = (c0 < CIN) ? (float)Arow[c0] : 0.f;
            float x1 = (c1 < CIN) ? (float)Arow[c1] : 0.f;
            if (BN) {
                if (c0 < CIN) x0 = fmaxf(fmaf(x0, ssc[c0], ssh[c0]), 0.f);
                if (c1 < CIN) x1 = fmaxf(fmaf(x1, ssc[c1], ssh[c1]), 0.f);
            }
            a[j] = (f16)x0;
            a[j + 8] = (f16)x1;
        }
        afrag[kb] = a;
    }

#pragma unroll
    for (int nt = 0; nt < O / 16; ++nt) {
        int ob = nt * 16;
        v8f acc = {0.f, 0.f, 0.f, 0.f, 0.f, 0.f, 0.f, 0.f};
#pragma unroll
        for (int kb = 0; kb < KBLK; ++kb) {
            // B fragment: col N = lane%16, halfs = K (same K order as A)
            const f16* bcol = &swT[(kb * 32) * O + ob + (lane & 15)];
            v16h bf;
#pragma unroll
            for (int j = 0; j < 8; ++j) { bf[j] = bcol[(hsel + j) * O]; bf[j + 8] = bcol[(16 + hsel + j) * O]; }
            acc = __builtin_amdgcn_wmma_f32_16x16x32_f16(
                false, afrag[kb], false, bf, (short)0, acc, false, false);
        }
        // C/D layout: VGPR r -> row m0 + hsel + r, col = ob + lane%16.
        // All 8 accumulators of one lane share a single output channel n.
        int n = ob + (lane & 15);
        int rbase = m0 + hsel;
        float s = 0.f, sq = 0.f;
#pragma unroll
        for (int r = 0; r < 8; ++r) {
            float v = acc[r];
            Hout[(size_t)(rbase + r) * O + n] = (f16)v;
            s += v; sq += v * v;
        }
        atomicAdd(&ssum[n], s);
        atomicAdd(&ssq[n], sq);
    }
    __syncthreads();
    if (t < O) { atomicAdd(&gsum[t], ssum[t]); atomicAdd(&gsq[t], ssq[t]); }
}

// ---------------------------------------------------------------------------
// 7. BN finalize: sums -> scale/shift. (Bias cancels under BN, so omitted.)
// ---------------------------------------------------------------------------
__global__ void bn_finalize_kernel(const float* __restrict__ sum, const float* __restrict__ sq,
                                   const float* __restrict__ g, const float* __restrict__ be,
                                   float* __restrict__ scale, float* __restrict__ shift, int O)
{
    int o = blockIdx.x * blockDim.x + threadIdx.x;
    if (o >= O) return;
    const float cnt = (float)M_TOT;
    float mean = sum[o] / cnt;
    float var  = sq[o] / cnt - mean * mean;
    float sc   = g[o] * rsqrtf(var + EPSV);
    scale[o] = sc;
    shift[o] = be[o] - mean * sc;
}

// ---------------------------------------------------------------------------
// 8. Apply BN3 + ReLU, max over the 32 samples, write transposed new_points.
// ---------------------------------------------------------------------------
__global__ void max_pool_kernel(const f16* __restrict__ h3,
                                const float* __restrict__ sc3, const float* __restrict__ sh3,
                                float* __restrict__ out)
{
    int i = blockIdx.x * blockDim.x + threadIdx.x;  // < B*S*128
    int o = i & 127;
    int q = i >> 7;
    int b = q / S_, s = q % S_;
    const f16* hp = h3 + (size_t)q * K_ * 128 + o;
    float a = sc3[o], c = sh3[o];
    float mx = 0.f;                                  // relu => max >= 0
    for (int k = 0; k < K_; ++k) {
        float v = (float)hp[k * 128];
        mx = fmaxf(mx, fmaf(v, a, c));
    }
    out[(size_t)(B_ * 3 * S_) + (size_t)b * 128 * S_ + (size_t)o * S_ + s] = mx;
}

// ---------------------------------------------------------------------------
extern "C" void kernel_launch(void* const* d_in, const int* in_sizes, int n_in,
                              void* d_out, int out_size, void* d_ws, size_t ws_size,
                              hipStream_t stream)
{
    const float* xyz    = (const float*)d_in[0];
    const float* points = (const float*)d_in[1];
    const float* w1  = (const float*)d_in[2];
    const float* g1  = (const float*)d_in[4];
    const float* be1 = (const float*)d_in[5];
    const float* w2  = (const float*)d_in[6];
    const float* g2  = (const float*)d_in[8];
    const float* be2 = (const float*)d_in[9];
    const float* w3  = (const float*)d_in[10];
    const float* g3  = (const float*)d_in[12];
    const float* be3 = (const float*)d_in[13];
    (void)in_sizes; (void)n_in; (void)out_size; (void)ws_size;

    char* wsp = (char*)d_ws;
    size_t off = 0;
    auto alloc = [&](size_t bytes) -> char* {
        char* p = wsp + off;
        off += (bytes + 255) & ~(size_t)255;
        return p;
    };
    int*   fidx   = (int*)  alloc((size_t)B_ * S_ * 4);
    float* newxyz = (float*)alloc((size_t)B_ * S_ * 3 * 4);
    int*   nidx   = (int*)  alloc((size_t)B_ * S_ * K_ * 4);
    f16*   wT1    = (f16*)  alloc(16 * 64 * 2);
    f16*   wT2    = (f16*)  alloc(64 * 64 * 2);
    f16*   wT3    = (f16*)  alloc(64 * 128 * 2);
    float* sums   = (float*)alloc(512 * 4);
    float* sum1 = sums,       *sq1 = sums + 64;
    float* sum2 = sums + 128, *sq2 = sums + 192;
    float* sum3 = sums + 256, *sq3 = sums + 384;
    float* bnp    = (float*)alloc(512 * 4);
    float* sc1 = bnp,       *sh1 = bnp + 64;
    float* sc2 = bnp + 128, *sh2 = bnp + 192;
    float* sc3 = bnp + 256, *sh3 = bnp + 384;
    f16* feat = (f16*)alloc((size_t)M_TOT * 16 * 2);
    f16* h1   = (f16*)alloc((size_t)M_TOT * 64 * 2);
    f16* h2   = (f16*)alloc((size_t)M_TOT * 64 * 2);
    f16* h3   = (f16*)alloc((size_t)M_TOT * 128 * 2);

    float* dout = (float*)d_out;

    zero_kernel<<<2, 256, 0, stream>>>(sums, 512);
    prep_wt_kernel<<<(16 * 64 + 255) / 256, 256, 0, stream>>>(w1, wT1, 64, 9, 16);
    prep_wt_kernel<<<(64 * 64 + 255) / 256, 256, 0, stream>>>(w2, wT2, 64, 64, 64);
    prep_wt_kernel<<<(64 * 128 + 255) / 256, 256, 0, stream>>>(w3, wT3, 128, 64, 64);

    fps_kernel<<<B_, 1024, 0, stream>>>(xyz, fidx);
    gather_newxyz_kernel<<<(B_ * S_ + 255) / 256, 256, 0, stream>>>(xyz, fidx, newxyz, dout);
    ballquery_kernel<<<B_ * S_ / 8, 256, 0, stream>>>(xyz, newxyz, nidx);
    build_feat_kernel<<<M_TOT / 256, 256, 0, stream>>>(xyz, points, newxyz, nidx, feat);

    gemm_wmma_kernel<16, 64, false><<<M_TOT / 128, 256, 0, stream>>>(
        feat, wT1, nullptr, nullptr, h1, sum1, sq1);
    bn_finalize_kernel<<<1, 64, 0, stream>>>(sum1, sq1, g1, be1, sc1, sh1, 64);

    gemm_wmma_kernel<64, 64, true><<<M_TOT / 128, 256, 0, stream>>>(
        h1, wT2, sc1, sh1, h2, sum2, sq2);
    bn_finalize_kernel<<<1, 64, 0, stream>>>(sum2, sq2, g2, be2, sc2, sh2, 64);

    gemm_wmma_kernel<64, 128, true><<<M_TOT / 128, 256, 0, stream>>>(
        h2, wT3, sc2, sh2, h3, sum3, sq3);
    bn_finalize_kernel<<<1, 128, 0, stream>>>(sum3, sq3, g3, be3, sc3, sh3, 128);

    max_pool_kernel<<<(B_ * S_ * 128) / 256, 256, 0, stream>>>(h3, sc3, sh3, dout);
}
